// SlidingWindowAttn_45234595561844
// MI455X (gfx1250) — matrix operand order repaired and checked by reading, not compiled, not run on hardware
//
#include <hip/hip_runtime.h>
#include <hip/hip_bf16.h>
#include <stdint.h>

typedef __attribute__((ext_vector_type(16))) __bf16 v16bf;
typedef __attribute__((ext_vector_type(8)))  float  v8f;

#define S_LEN   2048
#define DMODEL  2048
#define DKH     64
#define NQH     32
#define NKH     8
#define NB      2
#define WIN     64

union Frag16 { uint4 u[2]; v16bf v; };

__device__ __forceinline__ __bf16 to_bf16(float f) { return (__bf16)f; }

__device__ __forceinline__ v8f wmma_bf16(v16bf a, v16bf b, v8f c) {
  return __builtin_amdgcn_wmma_f32_16x16x32_bf16(false, a, false, b, (short)0, c, false, false);
}

// CDNA5 async memory->LDS (no VGPR staging), tracked by ASYNCcnt.
__device__ __forceinline__ void async_copy_b128(void* lds_ptr, const void* gptr) {
  uint32_t lds_off = (uint32_t)(uintptr_t)lds_ptr;  // low 32 bits of generic = LDS byte addr
  asm volatile("global_load_async_to_lds_b128 %0, %1, off"
               :: "v"(lds_off), "v"(gptr) : "memory");
}
__device__ __forceinline__ void async_wait0() {
  asm volatile("s_wait_asynccnt 0x0" ::: "memory");
}

// ------------------------------------------------------------- converts
// Tiled transpose + fp32->bf16: src (K x N) fp32 -> dst (N x K) bf16.
__global__ __launch_bounds__(256) void transpose_w_kernel(
    const float* __restrict__ src, __bf16* __restrict__ dst, int K, int N) {
  __shared__ float tile[32][33];
  int kb = blockIdx.y * 32, nb = blockIdx.x * 32;
  int tx = threadIdx.x, ty = threadIdx.y;  // (32, 8)
#pragma unroll
  for (int r = ty; r < 32; r += 8)
    tile[r][tx] = src[(size_t)(kb + r) * N + nb + tx];
  __syncthreads();
#pragma unroll
  for (int r = ty; r < 32; r += 8)
    dst[(size_t)(nb + r) * K + kb + tx] = to_bf16(tile[tx][r]);
}

// V (per head, S x 64 bf16) -> Vt (per head, 64 x S bf16)
__global__ __launch_bounds__(256) void transpose_v_kernel(
    const __bf16* __restrict__ src, __bf16* __restrict__ dst) {
  __shared__ __bf16 tile[32][33];
  int head = blockIdx.z;  // b*NKH + kh
  const __bf16* s0 = src + (size_t)head * S_LEN * DKH;
  __bf16* d0 = dst + (size_t)head * S_LEN * DKH;
  int sb = blockIdx.y * 32, db = blockIdx.x * 32;
  int tx = threadIdx.x, ty = threadIdx.y;  // (32, 8)
#pragma unroll
  for (int r = ty; r < 32; r += 8)
    tile[r][tx] = s0[(size_t)(sb + r) * DKH + db + tx];
  __syncthreads();
#pragma unroll
  for (int r = ty; r < 32; r += 8)
    d0[(size_t)(db + r) * S_LEN + sb + tx] = tile[tx][r];
}

__global__ __launch_bounds__(256) void embed_kernel(
    const int* __restrict__ X, const float* __restrict__ emb,
    __bf16* __restrict__ xb) {
  int bs = blockIdx.x;
  int tok = X[bs];
  const float* src = emb + (size_t)tok * DMODEL;
  __bf16* dst = xb + (size_t)bs * DMODEL;
  for (int i = threadIdx.x; i < DMODEL; i += 256) dst[i] = to_bf16(src[i]);
}

// ------------------------------------------------------------- GEMM
#define BM 128
#define BN 128
#define BKK 32
#define LDT 40   // 32 + pad; 80B row stride keeps every b128 16B-aligned

// C = A(M x Kd bf16 rowmajor) * Wt(N x Kd bf16 rowmajor, i.e. W transposed).
// MODE 0: bf16 out, head layout O[((b*H+h)*S + s)*64 + d], m=b*S+s, n=h*64+d
// MODE 1: fp32 out, row-major, + bias
template <int MODE>
__global__ __launch_bounds__(256) void gemm_kernel(
    const __bf16* __restrict__ A, const __bf16* __restrict__ Wt,
    void* __restrict__ Optr, const float* __restrict__ bias,
    int Kd, int N, int H) {
  __shared__ alignas(16) __bf16 As[2][BM * LDT];
  __shared__ alignas(16) __bf16 Bs[2][BN * LDT];
  const int tid = threadIdx.x;
  const int lane = tid & 31;
  const int wave = tid >> 5;
  const int mBase = blockIdx.y * BM;
  const int nBase = blockIdx.x * BN;

  v8f zero = {};
  v8f acc[8];
#pragma unroll
  for (int n = 0; n < 8; ++n) acc[n] = zero;

  auto issue = [&](int buf, int k0) {
#pragma unroll
    for (int c = 0; c < 2; ++c) {
      int cc = tid + c * 256;
      int row = cc >> 2;          // 0..127
      int ko  = (cc & 3) * 8;     // 0,8,16,24
      async_copy_b128(&As[buf][row * LDT + ko],
                      &A[(size_t)(mBase + row) * Kd + k0 + ko]);
      async_copy_b128(&Bs[buf][row * LDT + ko],
                      &Wt[(size_t)(nBase + row) * Kd + k0 + ko]);
    }
  };

  const int NT = Kd / BKK;
  issue(0, 0);
  for (int it = 0; it < NT; ++it) {
    async_wait0();
    __syncthreads();
    if (it + 1 < NT) issue((it + 1) & 1, (it + 1) * BKK);

    const __bf16* as = As[it & 1];
    const __bf16* bs = Bs[it & 1];
    Frag16 ta;
    {
      int rowa = lane & 15;
      int kb = (lane < 16) ? 0 : 8;
      const __bf16* ap = &as[(wave * 16 + rowa) * LDT + kb];
      ta.u[0] = *(const uint4*)(ap);
      ta.u[1] = *(const uint4*)(ap + 16);
    }
    int colb = lane & 15;
  int kb2  = (lane < 16) ? 0 : 8;
#pragma unroll
    for (int n = 0; n < 8; ++n) {
      Frag16 tb;
      const __bf16* bp = &bs[(n * 16 + colb) * LDT + kb2];
      tb.u[0] = *(const uint4*)(bp);
      tb.u[1] = *(const uint4*)(bp + 16);
      acc[n] = wmma_bf16(ta.v, tb.v, acc[n]);
    }
  }

  int col = lane & 15;
  int hi  = (lane < 16) ? 0 : 8;
#pragma unroll
  for (int n = 0; n < 8; ++n) {
#pragma unroll
    for (int i = 0; i < 8; ++i) {
      int m  = mBase + wave * 16 + i + hi;
      int nn = nBase + n * 16 + col;
      if constexpr (MODE == 0) {
        int bi = m >> 11;          // / S_LEN
        int s  = m & (S_LEN - 1);
        int h  = nn >> 6;
        int d  = nn & 63;
        ((__bf16*)Optr)[(((size_t)bi * H + h) * S_LEN + s) * DKH + d] =
            to_bf16(acc[n][i]);
      } else {
        ((float*)Optr)[(size_t)m * N + nn] = acc[n][i] + bias[nn];
      }
    }
  }
}

// ------------------------------------------------------------- attention
// One wave per (b, qh, 16-query tile). K read straight from global
// (row-major per head), V read straight from global transposed (Vt[d][s]).
// Out-of-window score columns are never read; out-of-window P entries are 0.
__global__ __launch_bounds__(32) void attn_kernel(
    const __bf16* __restrict__ Qb, const __bf16* __restrict__ Kb,
    const __bf16* __restrict__ Vt, __bf16* __restrict__ ctx) {
  __shared__ alignas(16) float  Sc[16 * 80];
  __shared__ alignas(16) __bf16 Ps[16 * 96];  // cols 80..95 zero

  const int lane = threadIdx.x & 31;
  const int q0 = blockIdx.x * 16;
  const int qh = blockIdx.y;
  const int b  = blockIdx.z;
  const int kh = qh >> 2;  // NQH/NKH = 4

  const __bf16* Qp  = Qb + (size_t)(b * NQH + qh) * S_LEN * DKH;
  const __bf16* Kp  = Kb + (size_t)(b * NKH + kh) * S_LEN * DKH;
  const __bf16* Vtp = Vt + (size_t)(b * NKH + kh) * DKH * S_LEN;  // [d][s]

  // Q fragments (DK=64 -> 2 K-steps), straight from global
  Frag16 qa[2];
  {
    int row = q0 + (lane & 15);
    int kb = (lane < 16) ? 0 : 8;
    const __bf16* qrow = Qp + (size_t)row * DKH;
#pragma unroll
    for (int t = 0; t < 2; ++t) {
      qa[t].u[0] = *(const uint4*)(qrow + t * 32 + kb);
      qa[t].u[1] = *(const uint4*)(qrow + t * 32 + 16 + kb);
    }
  }

  // scores: Q(16x64) x K^T(64x80); K fragments direct from global
  {
    int colk = lane & 15;
    int kb2 = (lane < 16) ? 0 : 8;
    int hi  = (lane < 16) ? 0 : 8;
#pragma unroll
    for (int n = 0; n < 5; ++n) {
      v8f acc = {};
      int key = q0 - WIN + n * 16 + colk;  // may be < 0: lands in valid ws mem,
                                           // column masked out of softmax
#pragma unroll
      for (int t = 0; t < 2; ++t) {
        Frag16 tb;
        const __bf16* kp = Kp + (ptrdiff_t)key * DKH + t * 32 + kb2;
        tb.u[0] = *(const uint4*)(kp);
        tb.u[1] = *(const uint4*)(kp + 16);
        acc = wmma_bf16(qa[t].v, tb.v, acc);
      }
#pragma unroll
      for (int i = 0; i < 8; ++i)
        Sc[(i + hi) * 80 + n * 16 + colk] = acc[i] * 0.125f;  // 1/sqrt(64)
    }
  }
  __syncthreads();

  // masked softmax, row r on lane r (valid j: max(r, 64-q0) <= j <= r+64)
  if (lane < 16) {
    int r = lane;
    int jlo = WIN - q0;
    if (jlo < r) jlo = r;
    int jhi = r + WIN;
    float mx = -3.0e38f;
    for (int j = jlo; j <= jhi; ++j) mx = fmaxf(mx, Sc[r * 80 + j]);
    float sum = 0.f;
    for (int j = jlo; j <= jhi; ++j) {
      float e = __expf(Sc[r * 80 + j] - mx);
      Sc[r * 80 + j] = e;
      sum += e;
    }
    float inv = 1.f / sum;
    for (int j = 0; j < 96; ++j) {
      float p = (j >= jlo && j <= jhi) ? Sc[r * 80 + j] * inv : 0.f;
      Ps[r * 96 + j] = to_bf16(p);
    }
  }
  __syncthreads();

  // ctx = P(16x96) x V(96x64); V fragments direct from global Vt[d][s]
  v8f cacc[4];
  {
    v8f z = {};
    cacc[0] = z; cacc[1] = z; cacc[2] = z; cacc[3] = z;
  }
  {
    int rowp = lane & 15;
    int kb  = (lane < 16) ? 0 : 8;
    int kb2 = (lane < 16) ? 0 : 8;
    int colv = lane & 15;
#pragma unroll
    for (int t = 0; t < 3; ++t) {
      Frag16 ta;
      const __bf16* pp = &Ps[rowp * 96 + t * 32 + kb];
      ta.u[0] = *(const uint4*)(pp);
      ta.u[1] = *(const uint4*)(pp + 16);
#pragma unroll
      for (int n = 0; n < 4; ++n) {
        Frag16 tb;
        const __bf16* vp =
            Vtp + (size_t)(n * 16 + colv) * S_LEN + (ptrdiff_t)(q0 - WIN) +
            t * 32 + kb2;  // keys beyond strip hit P==0 entries
        tb.u[0] = *(const uint4*)(vp);
        tb.u[1] = *(const uint4*)(vp + 16);
        cacc[n] = wmma_bf16(ta.v, tb.v, cacc[n]);
      }
    }
  }

  // store ctx in (b, s, qh*64+d) bf16 layout for the final GEMM
  {
    int col = lane & 15;
    int hi  = (lane < 16) ? 0 : 8;
#pragma unroll
    for (int n = 0; n < 4; ++n) {
#pragma unroll
      for (int i = 0; i < 8; ++i) {
        int srow = q0 + i + hi;
        int d = n * 16 + col;
        ctx[((size_t)b * S_LEN + srow) * DMODEL + qh * DKH + d] =
            to_bf16(cacc[n][i]);
      }
    }
  }
}

// ------------------------------------------------------------- launch
extern "C" void kernel_launch(void* const* d_in, const int* in_sizes, int n_in,
                              void* d_out, int out_size, void* d_ws, size_t ws_size,
                              hipStream_t stream) {
  (void)in_sizes; (void)n_in; (void)out_size; (void)ws_size;
  const int*   X   = (const int*)d_in[0];
  const float* emb = (const float*)d_in[1];
  const float* Wk  = (const float*)d_in[2];
  const float* Wv  = (const float*)d_in[3];
  const float* Wq  = (const float*)d_in[4];
  const float* Wo  = (const float*)d_in[5];
  const float* bo  = (const float*)d_in[6];
  float* out = (float*)d_out;

  char* ws = (char*)d_ws;
  size_t off = 0;
  auto walloc = [&](size_t bytes) {
    char* p = ws + off;
    off += (bytes + 255) & ~(size_t)255;
    return p;
  };
  __bf16* Wq_t = (__bf16*)walloc((size_t)DMODEL * DMODEL * 2);
  __bf16* Wk_t = (__bf16*)walloc((size_t)DMODEL * NKH * DKH * 2);
  __bf16* Wv_t = (__bf16*)walloc((size_t)DMODEL * NKH * DKH * 2);
  __bf16* Wo_t = (__bf16*)walloc((size_t)DMODEL * DMODEL * 2);
  __bf16* xb   = (__bf16*)walloc((size_t)NB * S_LEN * DMODEL * 2);
  __bf16* Qb   = (__bf16*)walloc((size_t)NB * NQH * S_LEN * DKH * 2);
  __bf16* Kb   = (__bf16*)walloc((size_t)NB * NKH * S_LEN * DKH * 2);
  __bf16* Vb   = (__bf16*)walloc((size_t)NB * NKH * S_LEN * DKH * 2);
  __bf16* Vt   = (__bf16*)walloc((size_t)NB * NKH * S_LEN * DKH * 2);
  __bf16* ctx  = (__bf16*)walloc((size_t)NB * S_LEN * DMODEL * 2);

  dim3 tb(32, 8);
  transpose_w_kernel<<<dim3(DMODEL / 32, DMODEL / 32), tb, 0, stream>>>(
      Wq, Wq_t, DMODEL, DMODEL);
  transpose_w_kernel<<<dim3((NKH * DKH) / 32, DMODEL / 32), tb, 0, stream>>>(
      Wk, Wk_t, DMODEL, NKH * DKH);
  transpose_w_kernel<<<dim3((NKH * DKH) / 32, DMODEL / 32), tb, 0, stream>>>(
      Wv, Wv_t, DMODEL, NKH * DKH);
  transpose_w_kernel<<<dim3(DMODEL / 32, DMODEL / 32), tb, 0, stream>>>(
      Wo, Wo_t, DMODEL, DMODEL);

  embed_kernel<<<NB * S_LEN, 256, 0, stream>>>(X, emb, xb);

  const int Mrows = NB * S_LEN;  // 4096
  gemm_kernel<0><<<dim3(DMODEL / BN, Mrows / BM), 256, 0, stream>>>(
      xb, Wq_t, Qb, nullptr, DMODEL, DMODEL, NQH);
  gemm_kernel<0><<<dim3((NKH * DKH) / BN, Mrows / BM), 256, 0, stream>>>(
      xb, Wk_t, Kb, nullptr, DMODEL, NKH * DKH, NKH);
  gemm_kernel<0><<<dim3((NKH * DKH) / BN, Mrows / BM), 256, 0, stream>>>(
      xb, Wv_t, Vb, nullptr, DMODEL, NKH * DKH, NKH);

  transpose_v_kernel<<<dim3(DKH / 32, S_LEN / 32, NB * NKH), tb, 0, stream>>>(
      Vb, Vt);

  attn_kernel<<<dim3(S_LEN / 16, NQH, NB), 32, 0, stream>>>(Qb, Kb, Vt, ctx);

  gemm_kernel<1><<<dim3(DMODEL / BN, Mrows / BM), 256, 0, stream>>>(
      ctx, Wo_t, out, bo, DMODEL, DMODEL, 0);
}
